// GCN_SSA_Block_62130996904364
// MI455X (gfx1250) — matrix operand order repaired and checked by reading, not compiled, not run on hardware
//
#include <hip/hip_runtime.h>

// GCN + ProbSparse-attention block for MI455X (gfx1250, wave32, WMMA + TDM).
// One workgroup (256 thr = 8 wave32) per batch element; the whole per-batch
// working set lives in LDS (~261 KB of the 320 KB WGP LDS). All K>=32 GEMMs
// run on v_wmma_f32_16x16x32_f16; the x tile is staged into LDS with the
// Tensor Data Mover (tensor_load_to_lds + s_wait_tensorcnt) when available.

typedef __attribute__((ext_vector_type(16))) _Float16 v16h;
typedef __attribute__((ext_vector_type(8)))  float    v8f;
typedef unsigned int u32x4 __attribute__((ext_vector_type(4)));
typedef int          i32x4 __attribute__((ext_vector_type(4)));
typedef int          i32x8 __attribute__((ext_vector_type(8)));
typedef _Float16 h16;

#ifndef HAVE_TDM
#if defined(__has_builtin)
#if __has_builtin(__builtin_amdgcn_tensor_load_to_lds) && \
    __has_builtin(__builtin_amdgcn_s_wait_tensorcnt)
#define HAVE_TDM 1
#else
#define HAVE_TDM 0
#endif
#else
#define HAVE_TDM 0
#endif
#endif

namespace {

constexpr int Bn  = 1024;   // batch
constexpr int C   = 64;     // channels / graph nodes
constexpr int HT  = 128;    // feature length
constexpr int NH  = 8;      // GCN hidden width
constexpr int U   = 25;     // top-k selected queries (5*ceil(ln 64))
constexpr int S   = 25;     // sampled keys per position
constexpr int NTHREADS = 256;

// ---------------- LDS layout (byte offsets into dynamic smem) --------------
constexpr int OFF_GW1  = 0;                        // 128x8  f32
constexpr int OFF_GB1  = OFF_GW1  + HT*NH*4;       // 8      f32
constexpr int OFF_GW2  = OFF_GB1  + NH*4;          // 8x128  f32
constexpr int OFF_GB2  = OFF_GW2  + NH*HT*4;       // 128    f32
constexpr int OFF_IDX  = OFF_GB2  + HT*4;          // 128x25 i32
constexpr int OFF_XH   = OFF_IDX  + HT*S*4;        // 64x128 f16  x (B operand)
constexpr int OFF_WA   = OFF_XH   + C*HT*2;        // 64x64  f16  current weight
constexpr int OFF_F    = OFF_WA   + C*C*2;         // 64x128 f32  projection f
constexpr int OFF_FNH  = OFF_F    + C*HT*4;        // 64x128 f16  L2-normalized f
constexpr int OFF_SIM  = OFF_FNH  + C*HT*2;        // 64x64  f32
constexpr int OFF_ADJH = OFF_SIM  + C*C*4;         // 64x64  f16  normalized adj
constexpr int OFF_G1   = OFF_ADJH + C*C*2;         // 64x8   f32
constexpr int OFF_H    = OFF_G1   + C*NH*4;        // 64x8   f32
constexpr int OFF_H2H  = OFF_H    + C*NH*4;        // 64x128 f16  h @ gw2
constexpr int OFF_QH   = OFF_H2H  + C*HT*2;        // 64x128 f16  queries
constexpr int OFF_KH   = OFF_QH   + C*HT*2;        // 64x128 f16  keys
constexpr int OFF_VH   = OFF_KH   + C*HT*2;        // 64x128 f16  values
constexpr int OFF_CTX  = OFF_VH   + C*HT*2;        // 64x128 f32  TDM x-stage / cumsum ctx
constexpr int OFF_QS   = OFF_CTX  + C*HT*4;        // 64x25  f32  qk_sample
constexpr int OFF_MV   = OFF_QS   + C*S*4;         // 64     f32  sparsity metric
constexpr int OFF_MT   = OFF_MV   + C*4;           // 32     i32  top-k indices
constexpr int OFF_QRED = OFF_MT   + 32*4;          // 32x128 f16  gathered queries (padded)
constexpr int OFF_SC   = OFF_QRED + 32*HT*2;       // 32x64  f32  scores
constexpr int OFF_ATT  = OFF_SC   + 32*C*4;        // 32x64  f16  softmax(scores)
constexpr int OFF_RS   = OFF_ATT  + 32*C*2;        // 64     f32  1/rowsum
constexpr int OFF_RL2  = OFF_RS   + C*4;           // 64     f32  1/l2norm
constexpr int OFF_UPD  = OFF_RL2  + C*4;           // 32x128 f32  attn @ V
constexpr int OFF_BIAS = OFF_UPD  + 32*HT*4;       // 3x64   f32  bq/bk/bv (keeps FLAT out of WMMA loop)
constexpr int SMEM_BYTES = OFF_BIAS + 3*C*4;       // ~261 KB < 320 KB WGP LDS

// ---------------- WMMA fragment helpers (A/B: contiguous k = 16*(lane/16) + e; C/D per ISA) ---------------
// A fragment: 16x32 f16, row-major source with leading dim `ld`.
__device__ __forceinline__ v16h ldsA16x32(const h16* a, int m0, int k0, int ld) {
  const int lane = threadIdx.x & 31;
  const int m  = m0 + (lane & 15);
  const int kh = (lane >> 4) * 16;
  v16h r;
#pragma unroll
  for (int e = 0; e < 8; ++e) {
    r[e]     = a[m * ld + k0 + kh + e];        // K = kh..kh+7
    r[e + 8] = a[m * ld + k0 + kh + 8 + e];    // K = kh+8..kh+15
  }
  return r;
}
// B fragment: 32x16 f16 from row-major KxN storage.
__device__ __forceinline__ v16h ldsB_kn(const h16* b, int k0, int n0, int ld) {
  const int lane = threadIdx.x & 31;
  const int n  = n0 + (lane & 15);
  const int kb = k0 + (lane >> 4) * 16;
  v16h r;
#pragma unroll
  for (int e = 0; e < 16; ++e) r[e] = b[(kb + e) * ld + n];
  return r;
}
// B fragment where logical B = src^T, src row-major NxK.
__device__ __forceinline__ v16h ldsB_nk(const h16* src, int k0, int n0, int ld) {
  const int lane = threadIdx.x & 31;
  const int n  = n0 + (lane & 15);
  const int kb = k0 + (lane >> 4) * 16;
  v16h r;
#pragma unroll
  for (int e = 0; e < 16; ++e) r[e] = src[n * ld + kb + e];
  return r;
}
__device__ __forceinline__ v8f wmma_f16(v16h a, v16h b, v8f c) {
  return __builtin_amdgcn_wmma_f32_16x16x32_f16(false, a, false, b,
                                                (short)0, c, false, false);
}
// C/D: 8 VGPRs, lane%16 = N, lane/16 selects M base 0/8.
__device__ __forceinline__ void stC_f32(float* d, v8f c, int m0, int n0, int ld) {
  const int lane = threadIdx.x & 31;
  const int n  = n0 + (lane & 15);
  const int mb = m0 + (lane >> 4) * 8;
#pragma unroll
  for (int r = 0; r < 8; ++r) d[(mb + r) * ld + n] = c[r];
}
__device__ __forceinline__ void stC_f16(h16* d, v8f c, int m0, int n0, int ld) {
  const int lane = threadIdx.x & 31;
  const int n  = n0 + (lane & 15);
  const int mb = m0 + (lane >> 4) * 8;
#pragma unroll
  for (int r = 0; r < 8; ++r) d[(mb + r) * ld + n] = (h16)c[r];
}

__global__ __launch_bounds__(NTHREADS)
void gcn_ssa_kernel(const float* __restrict__ x,
                    const float* __restrict__ wq, const float* __restrict__ bq,
                    const float* __restrict__ wk, const float* __restrict__ bk,
                    const float* __restrict__ wv, const float* __restrict__ bv,
                    const float* __restrict__ gw1, const float* __restrict__ gb1,
                    const float* __restrict__ gw2, const float* __restrict__ gb2,
                    const float* __restrict__ gamma_p,
                    const int*   __restrict__ idx,
                    float* __restrict__ out)
{
  extern __shared__ __align__(16) char smem[];
  float* sgw1 = (float*)(smem + OFF_GW1);
  float* sgb1 = (float*)(smem + OFF_GB1);
  float* sgw2 = (float*)(smem + OFF_GW2);
  float* sgb2 = (float*)(smem + OFF_GB2);
  int*   sidx = (int*)  (smem + OFF_IDX);
  h16*   xh   = (h16*)  (smem + OFF_XH);
  h16*   wA   = (h16*)  (smem + OFF_WA);
  float* F    = (float*)(smem + OFF_F);
  h16*   fnh  = (h16*)  (smem + OFF_FNH);
  float* SIM  = (float*)(smem + OFF_SIM);
  h16*   adjh = (h16*)  (smem + OFF_ADJH);
  float* G1   = (float*)(smem + OFF_G1);
  float* Hb   = (float*)(smem + OFF_H);
  h16*   h2h  = (h16*)  (smem + OFF_H2H);
  h16*   Qh   = (h16*)  (smem + OFF_QH);
  h16*   Kh   = (h16*)  (smem + OFF_KH);
  h16*   Vh   = (h16*)  (smem + OFF_VH);
  float* CTX  = (float*)(smem + OFF_CTX);
  float* QS   = (float*)(smem + OFF_QS);
  float* MV   = (float*)(smem + OFF_MV);
  int*   MT   = (int*)  (smem + OFF_MT);
  h16*   QRED = (h16*)  (smem + OFF_QRED);
  float* SC   = (float*)(smem + OFF_SC);
  h16*   ATT  = (h16*)  (smem + OFF_ATT);
  float* rsinv= (float*)(smem + OFF_RS);
  float* rl2  = (float*)(smem + OFF_RL2);
  float* UPD  = (float*)(smem + OFF_UPD);
  float* sbias= (float*)(smem + OFF_BIAS);

  const int b    = blockIdx.x;
  const int tid  = threadIdx.x;
  const int wave = tid >> 5;
  const float* xb = x + (size_t)b * C * HT;

  // ---- stage 0: stage constants + x into LDS (one HBM pass) ----
#if HAVE_TDM
  // Tensor Data Mover: DMA the 64x128 f32 x tile (32 KB, contiguous) into the
  // CTX staging buffer. D# per CDNA5 ISA: count=1,
  // type=2(image), data_size=2(4B), 1-row tile: tile_dim0=tensor_dim0=
  // tensor_dim0_stride=8192 elements. All operands wave-uniform -> SGPRs.
  if (wave == 0) {
    const unsigned long long ga = (unsigned long long)(uintptr_t)xb;
    const unsigned ldsb = (unsigned)(uintptr_t)(void*)CTX;  // LDS byte offset (low 32 bits)
    u32x4 g0;
    g0[0] = 1u;                                      // count=1 (user descriptor)
    g0[1] = ldsb;                                    // lds_addr
    g0[2] = (unsigned)(ga & 0xffffffffu);            // global_addr[31:0]
    g0[3] = (unsigned)((ga >> 32) & 0x01ffffffu)     // global_addr[56:32]
            | (2u << 30);                            // type=2
    i32x8 g1;
    g1[0] = (int)(2u << 16);                         // wg_mask=0, data_size=4B
    g1[1] = (int)(0x2000u << 16);                    // tensor_dim0[15:0]=8192
    g1[2] = (int)(1u << 16);                         // tensor_dim0 hi=0, tensor_dim1=1
    g1[3] = (int)(0x2000u << 16);                    // tile_dim0=8192
    g1[4] = 1;                                       // tile_dim1=1, tile_dim2=0
    g1[5] = 0x2000;                                  // tensor_dim0_stride=8192
    g1[6] = (int)(0x2000u << 16);                    // tensor_dim1_stride lo
    g1[7] = 0;                                       // tensor_dim1_stride hi
    i32x4 gz = {0, 0, 0, 0};
#if __clang_major__ >= 23
    i32x8 gz8 = {0, 0, 0, 0, 0, 0, 0, 0};
    __builtin_amdgcn_tensor_load_to_lds(g0, g1, gz, gz, gz8, 0);
#else
    __builtin_amdgcn_tensor_load_to_lds(g0, g1, gz, gz, 0);
#endif
    __builtin_amdgcn_s_wait_tensorcnt(0);
  }
#endif
  for (int i = tid; i < HT*NH; i += NTHREADS) sgw1[i] = gw1[i];
  if (tid < NH) sgb1[tid] = gb1[tid];
  for (int i = tid; i < NH*HT; i += NTHREADS) sgw2[i] = gw2[i];
  if (tid < HT) sgb2[tid] = gb2[tid];
  for (int i = tid; i < HT*S;  i += NTHREADS) sidx[i] = idx[i];
  if (tid < C) {                       // biases to LDS: no FLAT in WMMA loops
    sbias[0*C + tid] = bq[tid];
    sbias[1*C + tid] = bk[tid];
    sbias[2*C + tid] = bv[tid];
  }
  __syncthreads();
#if HAVE_TDM
  for (int i = tid; i < C*HT; i += NTHREADS) xh[i] = (h16)CTX[i];
#else
  for (int i = tid; i < C*HT; i += NTHREADS) xh[i] = (h16)xb[i];
#endif
  __syncthreads();

  const float* Wp[3]   = {wq, wk, wv};
  h16*         OutH[3] = {Qh, Kh, Vh};

  // =================== Q / K / V branches (sequential) ======================
  for (int br = 0; br < 3; ++br) {
    const float* W = Wp[br];

    for (int i = tid; i < C*C; i += NTHREADS) wA[i] = (h16)W[i];
    __syncthreads();

    // f = W(64x64) @ x(64x128) + bias   [WMMA, 32 tiles, K=64]
    for (int t = wave; t < 32; t += 8) {
      const int mt = t >> 3, nt = t & 7;
      v8f acc = {};
      for (int kk = 0; kk < C; kk += 32)
        acc = wmma_f16(ldsA16x32(wA, mt*16, kk, C),
                       ldsB_kn(xh, kk, nt*16, HT), acc);
      const int lane = threadIdx.x & 31;
      const int mb = mt*16 + (lane >> 4) * 8;
#pragma unroll
      for (int r = 0; r < 8; ++r) acc[r] += sbias[br*C + mb + r];
      stC_f32(F, acc, mt*16, nt*16, HT);
    }
    __syncthreads();

    // per-row sum (row_normalize) and L2 norm (cosine sim)
    if (tid < C) {
      float s = 0.f, sq = 0.f;
      for (int k2 = 0; k2 < HT; ++k2) { float v = F[tid*HT+k2]; s += v; sq += v*v; }
      rsinv[tid] = (fabsf(s) < 1e-30f) ? 0.f : 1.f / s;
      rl2[tid]   = 1.f / fmaxf(sqrtf(sq), 1e-8f);
    }
    __syncthreads();
    for (int i = tid; i < C*HT; i += NTHREADS)
      fnh[i] = (h16)(F[i] * rl2[i >> 7]);
    __syncthreads();

    // sim = fn @ fn^T   [WMMA, 16 tiles, K=128]
    for (int t = wave; t < 16; t += 8) {
      const int mt = t >> 2, nt = t & 3;
      v8f acc = {};
      for (int kk = 0; kk < HT; kk += 32)
        acc = wmma_f16(ldsA16x32(fnh, mt*16, kk, HT),
                       ldsB_nk(fnh, kk, nt*16, HT), acc);
      stC_f32(SIM, acc, mt*16, nt*16, C);
    }
    __syncthreads();

    // threshold + symmetrize (== elementwise max) + I, then row-normalize
    if (tid < C) {
      const int i = tid;
      float rsum = 0.f;
      for (int j = 0; j < C; ++j) {
        float a  = SIM[i*C+j] > 0.5f ? 1.f : 0.f;
        float at = SIM[j*C+i] > 0.5f ? 1.f : 0.f;
        rsum += fmaxf(a, at) + (i == j ? 1.f : 0.f);
      }
      const float inv = (rsum > 0.f) ? 1.f / rsum : 0.f;
      for (int j = 0; j < C; ++j) {
        float a  = SIM[i*C+j] > 0.5f ? 1.f : 0.f;
        float at = SIM[j*C+i] > 0.5f ? 1.f : 0.f;
        adjh[i*C+j] = (h16)((fmaxf(a, at) + (i == j ? 1.f : 0.f)) * inv);
      }
    }
    __syncthreads();

    // GCN narrow ops on VALU (N=8 < WMMA tile)
    for (int o = tid; o < C*NH; o += NTHREADS) {       // g1 = rownorm(f) @ gw1
      const int i = o >> 3, j = o & 7;
      float acc = 0.f;
      for (int k2 = 0; k2 < HT; ++k2) acc += F[i*HT+k2] * sgw1[k2*NH+j];
      G1[o] = acc * rsinv[i];
    }
    __syncthreads();
    for (int o = tid; o < C*NH; o += NTHREADS) {       // h = relu(adj@g1 + b1)
      const int i = o >> 3, j = o & 7;
      float acc = 0.f;
      for (int k2 = 0; k2 < C; ++k2) acc += (float)adjh[i*C+k2] * G1[k2*NH+j];
      Hb[o] = fmaxf(acc + sgb1[j], 0.f);
    }
    __syncthreads();
    for (int o = tid; o < C*HT; o += NTHREADS) {       // h2 = h @ gw2
      const int i = o >> 7, d = o & 127;
      float acc = 0.f;
#pragma unroll
      for (int j = 0; j < NH; ++j) acc += Hb[i*NH+j] * sgw2[j*HT+d];
      h2h[o] = (h16)acc;
    }
    __syncthreads();

    // branch out = adj(64x64) @ h2(64x128) + gb2   [WMMA, 32 tiles, K=64]
    h16* OB = OutH[br];
    for (int t = wave; t < 32; t += 8) {
      const int mt = t >> 3, nt = t & 7;
      v8f acc = {};
      for (int kk = 0; kk < C; kk += 32)
        acc = wmma_f16(ldsA16x32(adjh, mt*16, kk, C),
                       ldsB_kn(h2h, kk, nt*16, HT), acc);
      const int lane = threadIdx.x & 31;
      const float b2 = sgb2[nt*16 + (lane & 15)];
#pragma unroll
      for (int r = 0; r < 8; ++r) acc[r] += b2;
      stC_f16(OB, acc, mt*16, nt*16, HT);
    }
    __syncthreads();
  }

  // ========================= ProbSparse attention ===========================
  // warm the residual x rows back toward near caches for the final epilogue
  {
    const int i = tid * 32;                 // one 128B line per thread
    if (i < C*HT) __builtin_prefetch(xb + i, 0, 1);
  }

  // qk_sample[h][s] = sum_l Q[h][l] * K[h][idx[l][s]]   (gather, VALU)
  for (int o = tid; o < C*S; o += NTHREADS) {
    const int hh = o / S, ss = o % S;
    float acc = 0.f;
    for (int l = 0; l < HT; ++l)
      acc += (float)Qh[hh*HT+l] * (float)Kh[hh*HT + sidx[l*S+ss]];
    QS[o] = acc;
  }
  __syncthreads();
  if (tid < C) {                                  // M = max - sum/Lk
    float mx = -1e30f, sm = 0.f;
    for (int ss = 0; ss < S; ++ss) { float v = QS[tid*S+ss]; mx = fmaxf(mx, v); sm += v; }
    MV[tid] = mx - sm * (1.f / (float)HT);
  }
  __syncthreads();
  if (tid == 0) {                                 // top-25 of 64, serial select
    unsigned long long used = 0ull;
    for (int u = 0; u < U; ++u) {
      float best = -1e30f; int bi = 0;
      for (int i = 0; i < C; ++i) {
        if ((used >> i) & 1ull) continue;
        if (MV[i] > best) { best = MV[i]; bi = i; }
      }
      used |= 1ull << bi;
      MT[u] = bi;
    }
  }
  __syncthreads();

  for (int o = tid; o < 32*HT; o += NTHREADS) {   // gather q_red, pad to M=32
    const int u = o >> 7, d = o & 127;
    QRED[o] = (u < U) ? Qh[MT[u]*HT + d] : (h16)0.f;
  }
  __syncthreads();

  // scores = q_red(32x128) @ K^T * 1/sqrt(128)   [WMMA, 8 tiles, K=128]
  const float scale = 0.08838834764831845f;
  for (int t = wave; t < 8; t += 8) {
    const int mt = t >> 2, nt = t & 3;
    v8f acc = {};
    for (int kk = 0; kk < HT; kk += 32)
      acc = wmma_f16(ldsA16x32(QRED, mt*16, kk, HT),
                     ldsB_nk(Kh, kk, nt*16, HT), acc);
#pragma unroll
    for (int r = 0; r < 8; ++r) acc[r] *= scale;
    stC_f32(SC, acc, mt*16, nt*16, C);
  }
  __syncthreads();

  for (int o = tid; o < 32*C; o += NTHREADS) ATT[o] = (h16)0.f;   // pad rows
  __syncthreads();
  if (tid < U) {                                  // softmax over n=64
    const int u = tid;
    float mx = -1e30f;
    for (int n = 0; n < C; ++n) mx = fmaxf(mx, SC[u*C+n]);
    float sm = 0.f;
    for (int n = 0; n < C; ++n) { float e = __expf(SC[u*C+n] - mx); SC[u*C+n] = e; sm += e; }
    const float inv = 1.f / sm;
    for (int n = 0; n < C; ++n) ATT[u*C+n] = (h16)(SC[u*C+n] * inv);
  }
  __syncthreads();

  // upd = attn(32x64) @ V(64x128)   [WMMA, 16 tiles, K=64]
  for (int t = wave; t < 16; t += 8) {
    const int mt = t >> 3, nt = t & 7;
    v8f acc = {};
    for (int kk = 0; kk < C; kk += 32)
      acc = wmma_f16(ldsA16x32(ATT, mt*16, kk, C),
                     ldsB_kn(Vh, kk, nt*16, HT), acc);
    stC_f32(UPD, acc, mt*16, nt*16, HT);
  }
  __syncthreads();

  if (tid < HT) {                                 // context = cumsum(V, axis=n)
    float acc = 0.f;
    for (int n = 0; n < C; ++n) { acc += (float)Vh[n*HT + tid]; CTX[n*HT + tid] = acc; }
  }
  __syncthreads();
  for (int o = tid; o < U*HT; o += NTHREADS) {    // scatter updated rows
    const int u = o >> 7, d = o & 127;
    CTX[MT[u]*HT + d] = UPD[u*HT + d];
  }
  __syncthreads();

  const float g = gamma_p[0];
  float* ob = out + (size_t)b * C * HT;
  for (int i = tid; i < C*HT; i += NTHREADS) ob[i] = g * CTX[i] + xb[i];
}

} // namespace

extern "C" void kernel_launch(void* const* d_in, const int* in_sizes, int n_in,
                              void* d_out, int out_size, void* d_ws, size_t ws_size,
                              hipStream_t stream) {
  (void)in_sizes; (void)n_in; (void)d_ws; (void)ws_size; (void)out_size;
  const float* x    = (const float*)d_in[0];
  const float* wq   = (const float*)d_in[1];
  const float* bq   = (const float*)d_in[2];
  const float* wk   = (const float*)d_in[3];
  const float* bk   = (const float*)d_in[4];
  const float* wv   = (const float*)d_in[5];
  const float* bv   = (const float*)d_in[6];
  const float* gw1  = (const float*)d_in[7];
  const float* gb1  = (const float*)d_in[8];
  const float* gw2  = (const float*)d_in[9];
  const float* gb2  = (const float*)d_in[10];
  const float* gmma = (const float*)d_in[11];
  const int*   idx  = (const int*)d_in[12];
  float* out = (float*)d_out;

  gcn_ssa_kernel<<<dim3(Bn), dim3(NTHREADS), SMEM_BYTES, stream>>>(
      x, wq, bq, wk, bk, wv, bv, gw1, gb1, gw2, gb2, gmma, idx, out);
}
